// MultiHeadAttention_61572651155480
// MI455X (gfx1250) — matrix operand order repaired and checked by reading, not compiled, not run on hardware
//
#include <hip/hip_runtime.h>
#include <hip/hip_bf16.h>

// ---------------------------------------------------------------------------
// Transformer encoder layer (MHA + LN + FFN + LN) for gfx1250 (MI455X).
// All GEMMs run through v_wmma_f32_16x16x32_bf16 (fp32 accumulate).
// ---------------------------------------------------------------------------

typedef __attribute__((ext_vector_type(16))) __bf16        v16bf;
typedef __attribute__((ext_vector_type(8)))  float         v8f;
typedef __attribute__((ext_vector_type(4)))  float         f32x4;
typedef __attribute__((ext_vector_type(4)))  unsigned int  u32x4;
typedef __attribute__((ext_vector_type(2)))  unsigned int  u32x2;

#define HDIM 1024
#define SEQ  2048
#define NHEAD 16
#define HD   64

union Frag16 {
    v16bf          v;
    unsigned short us[16];
    u32x4          q[2];
};

__device__ __forceinline__ unsigned short f2bf(float f) {
    unsigned int u = __float_as_uint(f);
    unsigned int r = u + 0x7fffu + ((u >> 16) & 1u);   // round-to-nearest-even
    return (unsigned short)(r >> 16);
}
__device__ __forceinline__ unsigned int pack2bf(float a, float b) {
    return (unsigned int)f2bf(a) | ((unsigned int)f2bf(b) << 16);
}

// --------------------------- fp32 -> bf16 copy (vectorized) -----------------
__global__ void cvt_f32_bf16(const float* __restrict__ src,
                             unsigned short* __restrict__ dst, int n4) {
    int i = blockIdx.x * blockDim.x + threadIdx.x;
    if (i < n4) {
        f32x4 v = ((const f32x4*)src)[i];
        u32x2 p; p.x = pack2bf(v[0], v[1]); p.y = pack2bf(v[2], v[3]);
        ((u32x2*)dst)[i] = p;
    }
}

// ------------------- GEMM: out[t,o] = sum_k X[t,k]*W[o,k] + b[o] -----------
// X: [T,1024] bf16, W: [1024,1024] bf16 (torch Linear layout).
// Block = 256 threads (8 waves), tile 128 rows x 32 cols, K-step 64.
// Register-prefetch pipeline (k+1 tile) + global_prefetch (k+2 tile).
__global__ void gemm_bias_kernel(const unsigned short* __restrict__ X,
                                 const unsigned short* __restrict__ W,
                                 const float* __restrict__ bias,
                                 float* __restrict__ outF,
                                 unsigned short* __restrict__ outB,
                                 int relu)
{
    __shared__ __attribute__((aligned(16))) unsigned short sX[128 * 64]; // 16 KB
    __shared__ __attribute__((aligned(16))) unsigned short sW[32 * 64];  //  4 KB

    const int tid  = threadIdx.x;
    const int wave = tid >> 5;
    const int lane = tid & 31;
    const int h    = lane >> 4;
    const int ln   = lane & 15;
    const int mBase = blockIdx.x * 128;
    const int nBase = blockIdx.y * 32;

    // staging assignments (constant trip, no divergence)
    int xrow[4], xcol[4];
#pragma unroll
    for (int i = 0; i < 4; ++i) {
        int c = tid + i * 256;           // 1024 chunks of 8 bf16 (128x64)
        xrow[i] = c >> 3; xcol[i] = (c & 7) * 8;
    }
    const int wrow = tid >> 3, wcol = (tid & 7) * 8;   // 256 chunks (32x64)

    u32x4 gx[4], gw;
#pragma unroll
    for (int i = 0; i < 4; ++i)
        gx[i] = *(const u32x4*)&X[(mBase + xrow[i]) * HDIM + xcol[i]];
    gw = *(const u32x4*)&W[(nBase + wrow) * HDIM + wcol];

    v8f acc0 = {}, acc1 = {};

    for (int kb = 0; kb < HDIM; kb += 64) {
        // store staged tile
#pragma unroll
        for (int i = 0; i < 4; ++i)
            *(u32x4*)&sX[xrow[i] * 64 + xcol[i]] = gx[i];
        *(u32x4*)&sW[wrow * 64 + wcol] = gw;
        __syncthreads();

        // prefetch next tile into registers (loads in flight during WMMAs)
        if (kb + 64 < HDIM) {
#pragma unroll
            for (int i = 0; i < 4; ++i)
                gx[i] = *(const u32x4*)&X[(mBase + xrow[i]) * HDIM + kb + 64 + xcol[i]];
            gw = *(const u32x4*)&W[(nBase + wrow) * HDIM + kb + 64 + wcol];
        }
        // cache-prefetch the tile after that (global_prefetch_b8, no counters)
        if (kb + 128 < HDIM) {
#pragma unroll
            for (int i = 0; i < 4; ++i)
                __builtin_prefetch(&X[(mBase + xrow[i]) * HDIM + kb + 128 + xcol[i]], 0, 1);
            __builtin_prefetch(&W[(nBase + wrow) * HDIM + kb + 128 + wcol], 0, 1);
        }

#pragma unroll
        for (int ks = 0; ks < 64; ks += 32) {
            Frag16 a, b0, b1;
            const unsigned short* rp = &sX[(wave * 16 + ln) * 64 + ks];
            a.q[0] = *(const u32x4*)(rp + h * 16);
            a.q[1] = *(const u32x4*)(rp + h * 16 + 8);
            const unsigned short* wp0 = &sW[ln * 64 + ks + h * 16];
            b0.q[0] = *(const u32x4*)(wp0);
            b0.q[1] = *(const u32x4*)(wp0 + 8);
            const unsigned short* wp1 = &sW[(16 + ln) * 64 + ks + h * 16];
            b1.q[0] = *(const u32x4*)(wp1);
            b1.q[1] = *(const u32x4*)(wp1 + 8);
            acc0 = __builtin_amdgcn_wmma_f32_16x16x32_bf16(
                false, a.v, false, b0.v, (short)0, acc0, false, false);
            acc1 = __builtin_amdgcn_wmma_f32_16x16x32_bf16(
                false, a.v, false, b1.v, (short)0, acc1, false, false);
        }
        __syncthreads();
    }

    // C/D layout: VGPR r, lanes 0-15 -> M=r, lanes 16-31 -> M=8+r, N=ln
#pragma unroll
    for (int r = 0; r < 8; ++r) {
        int row = mBase + wave * 16 + h * 8 + r;
        int c0 = nBase + ln, c1 = nBase + 16 + ln;
        float v0 = acc0[r] + bias[c0];
        float v1 = acc1[r] + bias[c1];
        if (relu) { v0 = fmaxf(v0, 0.f); v1 = fmaxf(v1, 0.f); }
        if (outF) { outF[row * HDIM + c0] = v0; outF[row * HDIM + c1] = v1; }
        if (outB) { outB[row * HDIM + c0] = f2bf(v0);
                    outB[row * HDIM + c1] = f2bf(v1); }
    }
}

// ------------------------------ attention ----------------------------------
// Block = 128 threads (4 waves) per (batch, head, 16-query stripe).
// LDS: sS 16x2048 f32 | sP 16x2048 bf16 | sV 32x64 bf16 | reduce scratch.
// Phase 1: waves partition the key range. Phase 3: wave w owns head-dim slice w.
#define ATTN_SS_BYTES  (16 * SEQ * 4)
#define ATTN_SP_BYTES  (16 * SEQ * 2)
#define ATTN_SV_BYTES  (32 * HD * 2)
#define ATTN_RED_BYTES (128 * 4 + 16 * 4 + 16 * 4)
#define ATTN_LDS_BYTES (ATTN_SS_BYTES + ATTN_SP_BYTES + ATTN_SV_BYTES + ATTN_RED_BYTES)

__global__ void attn_kernel(const unsigned short* __restrict__ Qb,
                            const unsigned short* __restrict__ Kb,
                            const unsigned short* __restrict__ Vb,
                            unsigned short* __restrict__ ctxb)
{
    extern __shared__ char smem[];
    float*          sS     = (float*)smem;
    unsigned short* sP     = (unsigned short*)(smem + ATTN_SS_BYTES);
    unsigned short* sV     = (unsigned short*)(smem + ATTN_SS_BYTES + ATTN_SP_BYTES);
    float*          red    = (float*)(smem + ATTN_SS_BYTES + ATTN_SP_BYTES + ATTN_SV_BYTES);
    float*          rowmax = red + 128;
    float*          rowinv = red + 144;

    const int tid  = threadIdx.x;
    const int wave = tid >> 5;
    const int lane = tid & 31;
    const int h    = lane >> 4;
    const int ln   = lane & 15;
    const int qBase = blockIdx.x * 16;
    const int head  = blockIdx.y;
    const int tB    = blockIdx.z * SEQ;

    // ---- Phase 1: scores = Q*K^T/8 ; wave w covers keys [w*512, w*512+512) ----
    const unsigned short* qrow = Qb + (tB + qBase + ln) * HDIM + head * HD;
    const int kb0 = wave * (SEQ / 4);
    for (int kb = kb0; kb < kb0 + SEQ / 4; kb += 16) {
        v8f acc = {};
        const unsigned short* krow = Kb + (tB + kb + ln) * HDIM + head * HD;
        if (kb + 16 < kb0 + SEQ / 4)   // cache-prefetch next K rows
            __builtin_prefetch(krow + 16 * HDIM, 0, 1);
#pragma unroll
        for (int dd = 0; dd < HD; dd += 32) {
            Frag16 a, b;
            a.q[0] = *(const u32x4*)(qrow + dd + h * 16);
            a.q[1] = *(const u32x4*)(qrow + dd + h * 16 + 8);
            b.q[0] = *(const u32x4*)(krow + dd + h * 16);
            b.q[1] = *(const u32x4*)(krow + dd + h * 16 + 8);
            acc = __builtin_amdgcn_wmma_f32_16x16x32_bf16(
                false, a.v, false, b.v, (short)0, acc, false, false);
        }
#pragma unroll
        for (int r = 0; r < 8; ++r)
            sS[(h * 8 + r) * SEQ + kb + ln] = acc[r] * 0.125f;
    }
    __syncthreads();

    // ---- Phase 2: softmax; thread = (row, 256-col segment) -------------------
    {
        const int row = tid & 15, seg = tid >> 4;           // 16 rows x 8 segs
        const f32x4* p = (const f32x4*)&sS[row * SEQ + seg * 256];
        float mx = -3.0e38f;
        for (int i = 0; i < 64; ++i) {
            f32x4 v = p[i];
            mx = fmaxf(mx, fmaxf(fmaxf(v[0], v[1]), fmaxf(v[2], v[3])));
        }
        red[tid] = mx;
        __syncthreads();
        if (seg == 0) {
            float m = red[row];
#pragma unroll
            for (int j = 1; j < 8; ++j) m = fmaxf(m, red[j * 16 + row]);
            rowmax[row] = m;
        }
        __syncthreads();
        mx = rowmax[row];

        float s = 0.f;
        u32x2* op = (u32x2*)&sP[row * SEQ + seg * 256];
        for (int i = 0; i < 64; ++i) {
            f32x4 v = p[i];
            float e0 = __expf(v[0] - mx), e1 = __expf(v[1] - mx);
            float e2 = __expf(v[2] - mx), e3 = __expf(v[3] - mx);
            s += (e0 + e1) + (e2 + e3);
            u32x2 pk; pk.x = pack2bf(e0, e1); pk.y = pack2bf(e2, e3);
            op[i] = pk;
        }
        red[tid] = s;
        __syncthreads();
        if (seg == 0) {
            float t = red[row];
#pragma unroll
            for (int j = 1; j < 8; ++j) t += red[j * 16 + row];
            rowinv[row] = 1.0f / t;
        }
        __syncthreads();
    }

    // ---- Phase 3: ctx = P*V; wave w owns head-dim cols [w*16, w*16+16) -------
    {
        const int dt = wave;
        const int vr0 = tid >> 3,         vc0 = (tid & 7) * 8;        // chunk tid
        const int vr1 = (tid + 128) >> 3, vc1 = ((tid + 128) & 7) * 8;// chunk tid+128
        u32x4 t0 = *(const u32x4*)&Vb[(tB + vr0) * HDIM + head * HD + vc0];
        u32x4 t1 = *(const u32x4*)&Vb[(tB + vr1) * HDIM + head * HD + vc1];

        v8f acc = {};
        for (int kb = 0; kb < SEQ; kb += 32) {
            *(u32x4*)&sV[vr0 * HD + vc0] = t0;
            *(u32x4*)&sV[vr1 * HD + vc1] = t1;
            __syncthreads();
            if (kb + 32 < SEQ) {   // register-prefetch next V tile
                t0 = *(const u32x4*)&Vb[(tB + kb + 32 + vr0) * HDIM + head * HD + vc0];
                t1 = *(const u32x4*)&Vb[(tB + kb + 32 + vr1) * HDIM + head * HD + vc1];
            }
            if (kb + 64 < SEQ)     // cache-prefetch the tile after that
                __builtin_prefetch(&Vb[(tB + kb + 64 + vr0) * HDIM + head * HD + vc0], 0, 1);
            Frag16 a, b;
            const unsigned short* pp = &sP[ln * SEQ + kb];
            a.q[0] = *(const u32x4*)(pp + h * 16);
            a.q[1] = *(const u32x4*)(pp + h * 16 + 8);
#pragma unroll
            for (int j = 0; j < 16; ++j)
                b.us[j] = sV[(h * 16 + j) * HD + dt * 16 + ln];
            acc = __builtin_amdgcn_wmma_f32_16x16x32_bf16(
                false, a.v, false, b.v, (short)0, acc, false, false);
            __syncthreads();
        }

#pragma unroll
        for (int r = 0; r < 8; ++r)
            ctxb[(tB + qBase + h * 8 + r) * HDIM + head * HD + dt * 16 + ln] =
                f2bf(acc[r] * rowinv[h * 8 + r]);
    }
}

// ---------------------- residual add + LayerNorm (vectorized) ---------------
__global__ void ln_kernel(const float* __restrict__ a,
                          const float* __restrict__ res,
                          const float* __restrict__ g,
                          const float* __restrict__ be,
                          float* __restrict__ outF,
                          unsigned short* __restrict__ outB)
{
    __shared__ float rs[256], rs2[256];
    const int row = blockIdx.x, tid = threadIdx.x;
    const f32x4* a4 = (const f32x4*)(a + row * HDIM);
    const f32x4* r4 = (const f32x4*)(res + row * HDIM);
    f32x4 v = a4[tid] + r4[tid];                 // 256 threads x 4 = 1024
    float s  = (v[0] + v[1]) + (v[2] + v[3]);
    float s2 = (v[0]*v[0] + v[1]*v[1]) + (v[2]*v[2] + v[3]*v[3]);
    rs[tid] = s; rs2[tid] = s2;
    __syncthreads();
    for (int off = 128; off > 0; off >>= 1) {
        if (tid < off) { rs[tid] += rs[tid + off]; rs2[tid] += rs2[tid + off]; }
        __syncthreads();
    }
    float mean = rs[0] * (1.0f / HDIM);
    float var  = rs2[0] * (1.0f / HDIM) - mean * mean;
    float rinv = rsqrtf(var + 1e-5f);
    f32x4 vg = ((const f32x4*)g)[tid], vb = ((const f32x4*)be)[tid];
    f32x4 y = (v - mean) * rinv * vg + vb;
    if (outF) ((f32x4*)(outF + row * HDIM))[tid] = y;
    if (outB) {
        u32x2 pk; pk.x = pack2bf(y[0], y[1]); pk.y = pack2bf(y[2], y[3]);
        ((u32x2*)(outB + row * HDIM))[tid] = pk;
    }
}

// ---------------------------------------------------------------------------
extern "C" void kernel_launch(void* const* d_in, const int* in_sizes, int n_in,
                              void* d_out, int out_size, void* d_ws, size_t ws_size,
                              hipStream_t stream) {
    (void)in_sizes; (void)n_in; (void)out_size; (void)ws_size;
    const float* x  = (const float*)d_in[0];
    const float* Wq = (const float*)d_in[1];  const float* bq = (const float*)d_in[2];
    const float* Wk = (const float*)d_in[3];  const float* bk = (const float*)d_in[4];
    const float* Wv = (const float*)d_in[5];  const float* bv = (const float*)d_in[6];
    const float* Wo = (const float*)d_in[7];  const float* bo = (const float*)d_in[8];
    const float* W1 = (const float*)d_in[9];  const float* b1 = (const float*)d_in[10];
    const float* W2 = (const float*)d_in[11]; const float* b2 = (const float*)d_in[12];
    const float* g1 = (const float*)d_in[13]; const float* be1 = (const float*)d_in[14];
    const float* g2 = (const float*)d_in[15]; const float* be2 = (const float*)d_in[16];

    const int T = 2 * SEQ;               // 4096 tokens
    const size_t MB = 1024 * 1024;
    char* ws = (char*)d_ws;
    unsigned short* xb   = (unsigned short*)(ws + 0 * MB);   // 8 MB  [T,H] bf16
    unsigned short* Wqb  = (unsigned short*)(ws + 8 * MB);   // 2 MB each
    unsigned short* Wkb  = (unsigned short*)(ws + 10 * MB);
    unsigned short* Wvb  = (unsigned short*)(ws + 12 * MB);
    unsigned short* Wob  = (unsigned short*)(ws + 14 * MB);
    unsigned short* W1b  = (unsigned short*)(ws + 16 * MB);
    unsigned short* W2b  = (unsigned short*)(ws + 18 * MB);
    unsigned short* Qb   = (unsigned short*)(ws + 20 * MB);  // 8 MB
    unsigned short* Kb   = (unsigned short*)(ws + 28 * MB);  // 8 MB
    unsigned short* Vb   = (unsigned short*)(ws + 36 * MB);  // 8 MB
    unsigned short* ctxb = (unsigned short*)(ws + 44 * MB);  // 8 MB
    float*          tmpF = (float*)(ws + 52 * MB);           // 16 MB [T,H] f32
    float*          ln1F = (float*)(ws + 68 * MB);           // 16 MB
    unsigned short* ln1B = (unsigned short*)(ws + 84 * MB);  // 8 MB
    unsigned short* h1b  = (unsigned short*)(ws + 92 * MB);  // 8 MB

    // 1) bf16 conversions (vectorized: one f32x4 per thread)
    cvt_f32_bf16<<<(T * HDIM / 4) / 256, 256, 0, stream>>>(x,  xb,  T * HDIM / 4);
    cvt_f32_bf16<<<(HDIM * HDIM / 4) / 256, 256, 0, stream>>>(Wq, Wqb, HDIM * HDIM / 4);
    cvt_f32_bf16<<<(HDIM * HDIM / 4) / 256, 256, 0, stream>>>(Wk, Wkb, HDIM * HDIM / 4);
    cvt_f32_bf16<<<(HDIM * HDIM / 4) / 256, 256, 0, stream>>>(Wv, Wvb, HDIM * HDIM / 4);
    cvt_f32_bf16<<<(HDIM * HDIM / 4) / 256, 256, 0, stream>>>(Wo, Wob, HDIM * HDIM / 4);
    cvt_f32_bf16<<<(HDIM * HDIM / 4) / 256, 256, 0, stream>>>(W1, W1b, HDIM * HDIM / 4);
    cvt_f32_bf16<<<(HDIM * HDIM / 4) / 256, 256, 0, stream>>>(W2, W2b, HDIM * HDIM / 4);

    dim3 ggrid(T / 128, HDIM / 32);      // 32 x 32 blocks
    // 2) Q/K/V projections (bf16 out)
    gemm_bias_kernel<<<ggrid, 256, 0, stream>>>(xb, Wqb, bq, nullptr, Qb, 0);
    gemm_bias_kernel<<<ggrid, 256, 0, stream>>>(xb, Wkb, bk, nullptr, Kb, 0);
    gemm_bias_kernel<<<ggrid, 256, 0, stream>>>(xb, Wvb, bv, nullptr, Vb, 0);

    // 3) attention: 4 waves / block, ~197 KB LDS / block
    attn_kernel<<<dim3(SEQ / 16, NHEAD, 2), 128, ATTN_LDS_BYTES, stream>>>(Qb, Kb, Vb, ctxb);

    // 4) output projection (f32) + residual + LN1 (f32 + bf16)
    gemm_bias_kernel<<<ggrid, 256, 0, stream>>>(ctxb, Wob, bo, tmpF, nullptr, 0);
    ln_kernel<<<T, 256, 0, stream>>>(tmpF, x, g1, be1, ln1F, ln1B);

    // 5) FFN: relu(fc1) -> fc2 -> residual + LN2 -> d_out
    gemm_bias_kernel<<<ggrid, 256, 0, stream>>>(ln1B, W1b, b1, nullptr, h1b, 1);
    gemm_bias_kernel<<<ggrid, 256, 0, stream>>>(h1b, W2b, b2, tmpF, nullptr, 0);
    ln_kernel<<<T, 256, 0, stream>>>(tmpF, ln1F, g2, be2, (float*)d_out, nullptr);
}
